// GnnActor_31482110280397
// MI455X (gfx1250) — compile-verified, matrix operand order checked by reading, not run on hardware
//
#include <hip/hip_runtime.h>

#define BATCH 65536
#define K1    288   // 281 padded to multiple of 32
#define H     256

typedef __attribute__((ext_vector_type(16))) _Float16 v16h;
typedef __attribute__((ext_vector_type(8)))  float    v8f;

union Frag {
    v16h v;
    struct { ulonglong2 lo, hi; } q;
};

// Load one 16-bit WMMA fragment (A or B operand) for wave32:
// lanes 0-15 hold K = kbase+0..7 and kbase+16..23 of row (lane&15);
// lanes 16-31 hold K = kbase+8..15 and kbase+24..31 (ISA 7.12.2 layout).
__device__ __forceinline__ v16h load_frag(const _Float16* rowPtr, int kbase, int lane) {
    const int kk = kbase + ((lane >> 4) << 3);
    const _Float16* p = rowPtr + kk;
    Frag f;
    f.q.lo = *(const ulonglong2*)(p);
    f.q.hi = *(const ulonglong2*)(p + 16);
    return f.v;
}

__device__ __forceinline__ v8f wmma_f16(v16h a, v16h b, v8f c) {
    return __builtin_amdgcn_wmma_f32_16x16x32_f16(false, a, false, b, (short)0, c, false, false);
}

// ---------------------------------------------------------------------------
// Weight prep: transpose fp32 weights into N-major, K-contiguous f16 arrays.
// ---------------------------------------------------------------------------
__global__ void k_prep(const float* __restrict__ w1, const float* __restrict__ w2,
                       const float* __restrict__ rw, const float* __restrict__ mw,
                       const float* __restrict__ lw,
                       _Float16* __restrict__ W1T, _Float16* __restrict__ W2T,
                       _Float16* __restrict__ RT,  _Float16* __restrict__ HT) {
    const int tid    = blockIdx.x * blockDim.x + threadIdx.x;
    const int stride = gridDim.x * blockDim.x;
    for (int i = tid; i < 256 * K1; i += stride) {
        int n = i / K1, k = i % K1;                 // phi_w1: (281,256) row-major
        W1T[i] = (k < 281) ? (_Float16)w1[k * 256 + n] : (_Float16)0.0f;
    }
    for (int i = tid; i < 256 * 256; i += stride) {
        int n = i >> 8, k = i & 255;
        W2T[i] = (_Float16)w2[k * 256 + n];
        RT[i]  = (_Float16)rw[k * 256 + n];
    }
    for (int i = tid; i < 16 * 256; i += stride) {  // heads: rows 0-3 mean, 4-7 logstd
        int n = i >> 8, k = i & 255;
        float v = (n < 4) ? mw[k * 4 + n] : ((n < 8) ? lw[k * 4 + (n - 4)] : 0.0f);
        HT[i] = (_Float16)v;
    }
}

// ---------------------------------------------------------------------------
// Kernel 1: per wave = 16 batch rows, 3 objects interleaved at the WMMA level
// so every weight fragment feeds 3 WMMAs.  Readout tile is produced per N-tile
// (no persistent 128-VGPR accumulator).  2 waves/block, 104448 B LDS/block.
// ---------------------------------------------------------------------------
__global__ void __launch_bounds__(64) k_phi(
    const float* __restrict__ obs, const float* __restrict__ edge,
    const float* __restrict__ b1,  const float* __restrict__ b2,
    const _Float16* __restrict__ W1T, const _Float16* __restrict__ W2T,
    _Float16* __restrict__ RD)
{
    extern __shared__ _Float16 lds[];
    const int lane = threadIdx.x & 31;
    const int w    = threadIdx.x >> 5;
    const int nl   = lane & 15;
    _Float16* xbuf = lds + (size_t)w * (3 * 16 * K1 + 3 * 16 * H); // 3 x (16x288)
    _Float16* hbuf = xbuf + 3 * 16 * K1;                           // 3 x (16x256)
    const int batchBase = blockIdx.x * 32 + w * 16;

    // ---- build X tiles for all 3 objects: [body | obj_o | max(e0,e1) | 0] ----
#pragma unroll
    for (int o = 0; o < 3; ++o) {
        const int E0[3] = {0, 1, 3};
        const int E1[3] = {2, 4, 5};
        _Float16* xo = xbuf + o * 16 * K1;
        for (int r = 0; r < 16; ++r) {
            const int b = batchBase + r;
            const float* obr = obs + (size_t)b * 55;
            const float* ea  = edge + ((size_t)E0[o] * BATCH + b) * 256;
            const float* eb  = edge + ((size_t)E1[o] * BATCH + b) * 256;
            for (int c = lane; c < K1; c += 32) {
                float v;
                if      (c < 10)  v = obr[c];
                else if (c < 25)  v = obr[10 + o * 15 + (c - 10)];
                else if (c < 281) v = fmaxf(ea[c - 25], eb[c - 25]);
                else              v = 0.0f;
                xo[r * K1 + c] = (_Float16)v;
            }
        }
    }

    // ---- layer 1: h_o = relu(X_o @ W1 + b1); B frag shared by 3 objects ----
#pragma unroll 2
    for (int j = 0; j < 16; ++j) {
        const float bv = b1[j * 16 + nl];
        v8f acc[3];
#pragma unroll
        for (int o = 0; o < 3; ++o)
#pragma unroll
            for (int i = 0; i < 8; ++i) acc[o][i] = bv;
        const _Float16* brow = W1T + (size_t)(j * 16 + nl) * K1;
#pragma unroll
        for (int k = 0; k < 9; ++k) {
            v16h bf = load_frag(brow, k * 32, lane);
#pragma unroll
            for (int o = 0; o < 3; ++o) {
                v16h a = load_frag(xbuf + (size_t)(o * 16 + nl) * K1, k * 32, lane);
                acc[o] = wmma_f16(a, bf, acc[o]);
            }
        }
#pragma unroll
        for (int o = 0; o < 3; ++o)
#pragma unroll
            for (int i = 0; i < 8; ++i) {
                const int m = i + ((lane >> 4) << 3);
                hbuf[(size_t)(o * 16 + m) * H + j * 16 + nl] =
                    (_Float16)fmaxf(acc[o][i], 0.0f);
            }
    }

    // ---- layer 2 + object-sum readout, written per N-tile ----
#pragma unroll 2
    for (int j = 0; j < 16; ++j) {
        const float bv = b2[j * 16 + nl];
        v8f acc[3];
#pragma unroll
        for (int o = 0; o < 3; ++o)
#pragma unroll
            for (int i = 0; i < 8; ++i) acc[o][i] = bv;
        const _Float16* brow = W2T + (size_t)(j * 16 + nl) * H;
#pragma unroll
        for (int k = 0; k < 8; ++k) {
            v16h bf = load_frag(brow, k * 32, lane);
#pragma unroll
            for (int o = 0; o < 3; ++o) {
                v16h a = load_frag(hbuf + (size_t)(o * 16 + nl) * H, k * 32, lane);
                acc[o] = wmma_f16(a, bf, acc[o]);
            }
        }
#pragma unroll
        for (int i = 0; i < 8; ++i) {
            const int m = i + ((lane >> 4) << 3);
            const float s = fmaxf(acc[0][i], 0.0f) + fmaxf(acc[1][i], 0.0f) +
                            fmaxf(acc[2][i], 0.0f);
            RD[(size_t)(batchBase + m) * 256 + j * 16 + nl] = (_Float16)s;
        }
    }
}

// ---------------------------------------------------------------------------
// Kernel 2: rho + heads; each wave owns 2 M-tiles so every B fragment feeds
// 2 WMMAs.  4 waves/block, 131072 B LDS/block.
// ---------------------------------------------------------------------------
__global__ void __launch_bounds__(128) k_rho(
    const _Float16* __restrict__ RD, const float* __restrict__ rb,
    const _Float16* __restrict__ RT, const _Float16* __restrict__ HT,
    const float* __restrict__ mb, const float* __restrict__ lb,
    float* __restrict__ out)
{
    extern __shared__ _Float16 lds[];
    const int lane = threadIdx.x & 31;
    const int w    = threadIdx.x >> 5;
    const int nl   = lane & 15;
    _Float16* rbuf = lds + (size_t)w * (4 * 16 * 256); // 2 input tiles
    _Float16* sbuf = rbuf + 2 * 16 * 256;              // 2 r-staging tiles
    const int batchBase = blockIdx.x * 128 + w * 32;

    // stage 2 readout tiles into LDS (16 B per lane, coalesced)
#pragma unroll
    for (int t = 0; t < 2; ++t)
        for (int r = 0; r < 16; ++r) {
            const ulonglong2* src =
                (const ulonglong2*)(RD + (size_t)(batchBase + t * 16 + r) * 256);
            ((ulonglong2*)(rbuf + (size_t)(t * 16 + r) * 256))[lane] = src[lane];
        }

    // ---- rho: r = relu(readout @ rho_w1 + rho_b1) ----
#pragma unroll 2
    for (int j = 0; j < 16; ++j) {
        const float bv = rb[j * 16 + nl];
        v8f acc[2];
#pragma unroll
        for (int t = 0; t < 2; ++t)
#pragma unroll
            for (int i = 0; i < 8; ++i) acc[t][i] = bv;
        const _Float16* brow = RT + (size_t)(j * 16 + nl) * 256;
#pragma unroll
        for (int k = 0; k < 8; ++k) {
            v16h bf = load_frag(brow, k * 32, lane);
#pragma unroll
            for (int t = 0; t < 2; ++t) {
                v16h a = load_frag(rbuf + (size_t)(t * 16 + nl) * 256, k * 32, lane);
                acc[t] = wmma_f16(a, bf, acc[t]);
            }
        }
#pragma unroll
        for (int t = 0; t < 2; ++t)
#pragma unroll
            for (int i = 0; i < 8; ++i) {
                const int m = i + ((lane >> 4) << 3);
                sbuf[(size_t)(t * 16 + m) * 256 + j * 16 + nl] =
                    (_Float16)fmaxf(acc[t][i], 0.0f);
            }
    }

    // ---- heads: one 16-col WMMA tile; cols 0-3 = mean, 4-7 = log_std ----
    const float hbv = (nl < 4) ? mb[nl] : ((nl < 8) ? lb[nl - 4] : 0.0f);
    v8f acc[2];
#pragma unroll
    for (int t = 0; t < 2; ++t)
#pragma unroll
        for (int i = 0; i < 8; ++i) acc[t][i] = hbv;
    const _Float16* brow = HT + (size_t)nl * 256;
#pragma unroll
    for (int k = 0; k < 8; ++k) {
        v16h bf = load_frag(brow, k * 32, lane);
#pragma unroll
        for (int t = 0; t < 2; ++t) {
            v16h a = load_frag(sbuf + (size_t)(t * 16 + nl) * 256, k * 32, lane);
            acc[t] = wmma_f16(a, bf, acc[t]);
        }
    }
#pragma unroll
    for (int t = 0; t < 2; ++t)
#pragma unroll
        for (int i = 0; i < 8; ++i) {
            const int m = i + ((lane >> 4) << 3);
            const int b = batchBase + t * 16 + m;
            const float v = acc[t][i];
            if (nl < 4)      out[(size_t)b * 4 + nl] = v;
            else if (nl < 8) out[(size_t)BATCH * 4 + (size_t)b * 4 + (nl - 4)] =
                                 fminf(fmaxf(v, -20.0f), 2.0f);
        }
}

// ---------------------------------------------------------------------------
extern "C" void kernel_launch(void* const* d_in, const int* in_sizes, int n_in,
                              void* d_out, int out_size, void* d_ws, size_t ws_size,
                              hipStream_t stream) {
    (void)in_sizes; (void)n_in; (void)out_size; (void)ws_size;
    const float* obs      = (const float*)d_in[0];
    const float* edge     = (const float*)d_in[1];
    const float* phi_w1   = (const float*)d_in[2];
    const float* phi_b1   = (const float*)d_in[3];
    const float* phi_w2   = (const float*)d_in[4];
    const float* phi_b2   = (const float*)d_in[5];
    const float* rho_w1   = (const float*)d_in[6];
    const float* rho_b1   = (const float*)d_in[7];
    const float* mean_w   = (const float*)d_in[8];
    const float* mean_b   = (const float*)d_in[9];
    const float* logstd_w = (const float*)d_in[10];
    const float* logstd_b = (const float*)d_in[11];

    char* ws = (char*)d_ws;
    _Float16* W1T = (_Float16*)(ws + 0);        // 256*288*2 = 147456
    _Float16* W2T = (_Float16*)(ws + 147456);   // 131072
    _Float16* RT  = (_Float16*)(ws + 278528);   // 131072
    _Float16* HT  = (_Float16*)(ws + 409600);   // 8192
    _Float16* RD  = (_Float16*)(ws + 417792);   // 65536*256*2 = 33.5 MB

    const int lds1 = (3 * 16 * K1 + 3 * 16 * H) * 2 * (int)sizeof(_Float16); // 104448
    const int lds2 = (8 * 16 * 256) * (int)sizeof(_Float16);                 // 65536 *2 waves-> see below
    const int lds2b = 2 * lds2;                                              // 131072

    // allow >64KB dynamic LDS (not a stream op; safe under graph capture)
    (void)hipFuncSetAttribute((const void*)k_phi,
                              hipFuncAttributeMaxDynamicSharedMemorySize, lds1);
    (void)hipFuncSetAttribute((const void*)k_rho,
                              hipFuncAttributeMaxDynamicSharedMemorySize, lds2b);

    k_prep<<<dim3(256), dim3(256), 0, stream>>>(phi_w1, phi_w2, rho_w1, mean_w, logstd_w,
                                                W1T, W2T, RT, HT);

    k_phi<<<dim3(BATCH / 32), dim3(64), lds1, stream>>>(obs, edge, phi_b1, phi_b2,
                                                        W1T, W2T, RD);

    k_rho<<<dim3(BATCH / 128), dim3(128), lds2b, stream>>>(RD, rho_b1, RT, HT,
                                                           mean_b, logstd_b,
                                                           (float*)d_out);
}
